// Trans4map_segformer_17832704213129
// MI455X (gfx1250) — matrix operand order repaired and checked by reading, not compiled, not run on hardware
//
#include <hip/hip_runtime.h>

// ---------------------------------------------------------------------------
// Trans4map-style pipeline for MI455X (gfx1250, wave32, WMMA).
//   - prefix-sum compaction of inlier mask (stable order)
//   - fused bilinear-upsample gather into 500x500x64 f16 map
//   - 4x (implicit-GEMM conv via v_wmma_f32_16x16x32_f16 -> BN(train) -> ReLU)
//     compile-time specialized; 32Mx32N wave tile, 4 WMMAs / wave / K-chunk
//   - final 1x1 conv + bias -> NCHW f32 output
// ---------------------------------------------------------------------------

typedef __attribute__((ext_vector_type(16))) _Float16     v16h;
typedef __attribute__((ext_vector_type(8)))  float        v8f;
typedef __attribute__((ext_vector_type(8)))  unsigned int v8u;

#define MAPW   500
#define HWMAP  (MAPW * MAPW)          // 250000 map cells
#define INH    1024
#define INW    2048
#define NPIX   (INH * INW)            // 2097152 interpolated pixels
#define SRCH   256
#define SRCW   512
#define CMEM   64
#define SCANB  512
#define NBLK   (NPIX / SCANB)         // 4096
#define STATS_SLICES 32

// ---------------------------------------------------------------------------
// 1) mask compaction: block sums -> single-block scan -> rank scatter
// ---------------------------------------------------------------------------
__global__ void mask_block_sums(const int* __restrict__ mask, int* __restrict__ bsum) {
    __shared__ int sh[SCANB];
    const int tid = threadIdx.x;
    sh[tid] = mask[blockIdx.x * SCANB + tid] ? 1 : 0;
    __syncthreads();
    for (int o = SCANB / 2; o > 0; o >>= 1) {
        if (tid < o) sh[tid] += sh[tid + o];
        __syncthreads();
    }
    if (tid == 0) bsum[blockIdx.x] = sh[0];
}

__global__ void scan_sums4096(int* __restrict__ bsum) {
    __shared__ int sh[1024];
    const int tid = threadIdx.x;
    int carry = 0;
    for (int ch = 0; ch < NBLK / 1024; ++ch) {
        const int i = ch * 1024 + tid;
        sh[tid] = bsum[i];
        __syncthreads();
        for (int o = 1; o < 1024; o <<= 1) {
            const int t = (tid >= o) ? sh[tid - o] : 0;
            __syncthreads();
            sh[tid] += t;
            __syncthreads();
        }
        bsum[i] = sh[tid] + carry;
        __syncthreads();
        carry += sh[1023];
        __syncthreads();
    }
}

__global__ void scatter_pos(const int* __restrict__ mask, const int* __restrict__ bsum,
                            int* __restrict__ pos) {
    __shared__ int sh[SCANB];
    const int tid = threadIdx.x;
    const int p   = blockIdx.x * SCANB + tid;
    const int m   = mask[p] ? 1 : 0;
    sh[tid] = m;
    __syncthreads();
    for (int o = 1; o < SCANB; o <<= 1) {
        const int t = (tid >= o) ? sh[tid - o] : 0;
        __syncthreads();
        sh[tid] += t;
        __syncthreads();
    }
    const int offset = blockIdx.x ? bsum[blockIdx.x - 1] : 0;
    const int n_in   = bsum[NBLK - 1];
    const int incl   = sh[tid] + offset;        // #inliers in [0..p]
    if (m) pos[incl - 1]        = p;            // stable inlier rank
    else   pos[n_in + p - incl] = p;            // stable outlier rank
}

// ---------------------------------------------------------------------------
// 2) sentinel = max(proj_indices)
// ---------------------------------------------------------------------------
__global__ void init_max(int* d) { if (threadIdx.x == 0) d[0] = 0; }

__global__ void max_reduce(const int* __restrict__ idx, int n, int* __restrict__ d) {
    __shared__ int sh[256];
    int local = 0;
    for (int i = blockIdx.x * blockDim.x + threadIdx.x; i < n; i += gridDim.x * blockDim.x)
        local = max(local, idx[i]);
    sh[threadIdx.x] = local;
    __syncthreads();
    for (int o = 128; o > 0; o >>= 1) {
        if (threadIdx.x < o) sh[threadIdx.x] = max(sh[threadIdx.x], sh[threadIdx.x + o]);
        __syncthreads();
    }
    if (threadIdx.x == 0) atomicMax(d, sh[0]);
}

// ---------------------------------------------------------------------------
// 3) fused bilinear (align_corners) gather -> f16 memory map + rgb + observed
//    one wave per map cell (wave32: 2 channels / lane)
// ---------------------------------------------------------------------------
__global__ void gather_build(const float* __restrict__ feats, const int* __restrict__ idx,
                             const int* __restrict__ pos, const float* __restrict__ rgb,
                             const int* __restrict__ dmax, _Float16* __restrict__ actH,
                             float* __restrict__ obs, float* __restrict__ rgbw) {
    const int cell = blockIdx.x * 8 + (threadIdx.x >> 5);
    const int lane = threadIdx.x & 31;
    if (cell >= HWMAP) return;
    const int  j     = idx[cell];
    const bool valid = j < dmax[0];
    if (valid) {
        const int p  = pos[j];
        const int Y  = p >> 11;           // / 2048
        const int X  = p & 2047;
        const float ys = Y * ((float)(SRCH - 1) / (float)(INH - 1));
        const float xs = X * ((float)(SRCW - 1) / (float)(INW - 1));
        const int y0 = (int)ys, x0 = (int)xs;
        const int y1 = min(y0 + 1, SRCH - 1), x1 = min(x0 + 1, SRCW - 1);
        const float wy = ys - (float)y0, wx = xs - (float)x0;
        const float w00 = (1.f - wy) * (1.f - wx), w01 = (1.f - wy) * wx;
        const float w10 = wy * (1.f - wx),         w11 = wy * wx;
        for (int c = lane; c < CMEM; c += 32) {
            const float* fc = feats + (size_t)c * (SRCH * SRCW);
            const float v = w00 * fc[y0 * SRCW + x0] + w01 * fc[y0 * SRCW + x1] +
                            w10 * fc[y1 * SRCW + x0] + w11 * fc[y1 * SRCW + x1];
            actH[(size_t)cell * CMEM + c] = (_Float16)v;
        }
        if (lane < 3)  rgbw[cell * 3 + lane] = rgb[(size_t)p * 3 + lane];
        if (lane == 0) obs[cell] = 1.0f;
    } else {
        for (int c = lane; c < CMEM; c += 32) actH[(size_t)cell * CMEM + c] = (_Float16)0.f;
        if (lane < 3)  rgbw[cell * 3 + lane] = 0.f;
        if (lane == 0) obs[cell] = 0.f;
    }
}

// ---------------------------------------------------------------------------
// 4) weight pack: OIHW f32 -> [K = (ky*KW+kx)*CIN + ci][N = co] f16
// ---------------------------------------------------------------------------
__global__ void pack_w(const float* __restrict__ w, _Float16* __restrict__ wp,
                       int COUT, int CIN, int KH, int KW) {
    const int total = KH * KW * CIN * COUT;
    for (int i = blockIdx.x * blockDim.x + threadIdx.x; i < total;
         i += gridDim.x * blockDim.x) {
        const int n  = i % COUT;
        const int k  = i / COUT;
        const int c  = k % CIN;
        const int r  = k / CIN;
        const int kx = r % KW;
        const int ky = r / KW;
        wp[i] = (_Float16)w[(((size_t)n * CIN + c) * KH + ky) * KW + kx];
    }
}

// ---------------------------------------------------------------------------
// 5) implicit-GEMM conv: M = 250000 pixels, N = COUT, K = KH*KW*CIN.
//    Block = 256 thr (8 waves), tile 128(M) x 64(N), K chunks of 32 via LDS.
//    Each wave: 32M x 32N -> 2 A-frags x 2 B-frags -> 4 independent
//    v_wmma_f32_16x16x32_f16 per chunk (1.0 LDS-frag load per WMMA).
//    Fragments assembled in ext-vector registers + __builtin_bit_cast
//    (no union -> no SROA failure -> no private-memory round trip).
// ---------------------------------------------------------------------------
template <int CIN, int COUT, int KH, int KW, int PAD>
__global__ void __launch_bounds__(256, 1)
conv_wmma(const _Float16* __restrict__ act, const _Float16* __restrict__ wp,
          float* __restrict__ out) {
    constexpr int K = KH * KW * CIN;
    __shared__ __align__(16) _Float16 As[128 * 32];  // [pixel-row][k]
    __shared__ __align__(16) _Float16 Bs[64 * 32];   // [n][k] (transposed)

    const int tid  = threadIdx.x;
    const int lane = tid & 31;
    const int wave = tid >> 5;
    const int wM   = wave & 3;            // 4 waves along M (32 rows each)
    const int wN   = wave >> 2;           // 2 waves along N (32 cols each)
    const int blockM = blockIdx.x * 128;
    const int blockN = blockIdx.y * 64;

    v8f acc00 = {}, acc01 = {}, acc10 = {}, acc11 = {};

    // A-staging: 2 threads per pixel row; each fills 16 k-slots
    const int aRow = tid & 127;
    const int aKg  = (tid >> 7) * 16;
    const int pix  = blockM + aRow;
    const int py   = pix / MAPW;
    const int px   = pix % MAPW;
    const bool pv  = pix < HWMAP;

    // fragment source addresses in LDS (constant across chunks), uint units
    const int hp = (lane >> 4) * 4;                    // A pair offset
    const int hb = (lane >> 4) * 4;                    // B pair offset
    const unsigned int* Arow0 =
        (const unsigned int*)&As[((wM * 32) + (lane & 15)) * 32];
    const unsigned int* Arow1 = Arow0 + 16 * 16;       // +16 rows (16*32 halfs)
    const unsigned int* Brow0 =
        (const unsigned int*)&Bs[((wN * 32) + (lane & 15)) * 32];
    const unsigned int* Brow1 = Brow0 + 16 * 16;       // +16 rows

    for (int k0 = 0; k0 < K; k0 += 32) {
        // ---- prefetch next weight chunk (global_prefetch_b8) ----
        if (k0 + 32 < K)
            __builtin_prefetch(&wp[(size_t)(k0 + 32) * COUT + blockN] + tid, 0, 3);
        // ---- stage A chunk (im2col gather, zero-padded borders + K tail) ----
#pragma unroll
        for (int j = 0; j < 16; ++j) {
            const int kl = aKg + j;
            const int k  = k0 + kl;
            float vv = 0.f;
            if (pv && k < K) {
                const int c  = k % CIN;      // compile-time CIN -> cheap
                const int r  = k / CIN;
                const int kx = r % KW;
                const int ky = r / KW;
                const int yy = py + ky - PAD;
                const int xx = px + kx - PAD;
                if (yy >= 0 && yy < MAPW && xx >= 0 && xx < MAPW)
                    vv = (float)act[((size_t)(yy * MAPW + xx)) * CIN + c];
            }
            As[aRow * 32 + kl] = (_Float16)vv;
        }
        // ---- stage B chunk: Bs[n][k] = wp[(k0+k)*COUT + blockN+n] ----
#pragma unroll
        for (int i = 0; i < 8; ++i) {
            const int e = tid * 8 + i;
            const int n = e >> 5, k = e & 31;
            const int gk = k0 + k, gn = blockN + n;
            _Float16 vv = (_Float16)0.f;
            if (gk < K && gn < COUT) vv = wp[(size_t)gk * COUT + gn];
            Bs[n * 32 + k] = vv;
        }
        __syncthreads();

        // ---- build wave fragments per CDNA5 16-bit WMMA VGPR layouts ----
        // (vector-element inserts + bit_cast: stays entirely in VGPRs)
        v8u a0u = {}, a1u = {}, b0u = {}, b1u = {};
        // A 16x32: lanes 0-15 pairs {0..3, 8..11}; lanes 16-31 pairs {4..7, 12..15}
#pragma unroll
        for (int v = 0; v < 4; ++v) { a0u[v] = Arow0[v + hp]; a1u[v] = Arow1[v + hp]; }
#pragma unroll
        for (int v = 4; v < 8; ++v) { a0u[v] = Arow0[v + 4 + hp]; a1u[v] = Arow1[v + 4 + hp]; }
        // B 32x16: lane column N; lanes 0-15 K pairs {0..3, 8..11}; lanes 16-31 pairs {4..7, 12..15}
#pragma unroll
        for (int v = 0; v < 8; ++v) { b0u[v] = Brow0[v + (v >> 2) * 4 + hb]; b1u[v] = Brow1[v + (v >> 2) * 4 + hb]; }
        const v16h af0 = __builtin_bit_cast(v16h, a0u);
        const v16h af1 = __builtin_bit_cast(v16h, a1u);
        const v16h bf0 = __builtin_bit_cast(v16h, b0u);
        const v16h bf1 = __builtin_bit_cast(v16h, b1u);

        // 4 independent WMMAs (distinct accumulators -> no D->A/B hazards)
        acc00 = __builtin_amdgcn_wmma_f32_16x16x32_f16(
            false, af0, false, bf0, (short)0, acc00, false, false);
        acc01 = __builtin_amdgcn_wmma_f32_16x16x32_f16(
            false, af0, false, bf1, (short)0, acc01, false, false);
        acc10 = __builtin_amdgcn_wmma_f32_16x16x32_f16(
            false, af1, false, bf0, (short)0, acc10, false, false);
        acc11 = __builtin_amdgcn_wmma_f32_16x16x32_f16(
            false, af1, false, bf1, (short)0, acc11, false, false);
        __syncthreads();
    }

    // ---- store D: VGPR r -> M = r + (lane>=16 ? 8 : 0), N = lane & 15 ----
    const int mofs = (lane >> 4) * 8;
    const int mlo  = blockM + wM * 32 + mofs;
    const int nlo  = blockN + wN * 32 + (lane & 15);
#pragma unroll
    for (int r = 0; r < 8; ++r) {
        const int p0 = mlo + r;
        const int p1 = p0 + 16;
        if (nlo < COUT) {
            if (p0 < HWMAP) out[(size_t)p0 * COUT + nlo] = acc00[r];
            if (p1 < HWMAP) out[(size_t)p1 * COUT + nlo] = acc10[r];
        }
        if (nlo + 16 < COUT) {
            if (p0 < HWMAP) out[(size_t)p0 * COUT + nlo + 16] = acc01[r];
            if (p1 < HWMAP) out[(size_t)p1 * COUT + nlo + 16] = acc11[r];
        }
    }
}

// ---------------------------------------------------------------------------
// 6) training-mode BN: per-channel biased mean/var (parallel partials +
//    global_atomic_add_f32), then norm+ReLU -> f16
// ---------------------------------------------------------------------------
__global__ void zero_stats(float* __restrict__ sums, float* __restrict__ sumsq) {
    if (threadIdx.x < 128) {
        sums[threadIdx.x]  = 0.f;
        sumsq[threadIdx.x] = 0.f;
    }
}

__global__ void bn_partial(const float* __restrict__ x, int C,
                           float* __restrict__ sums, float* __restrict__ sumsq) {
    __shared__ float s1[256], s2[256];
    const int c     = blockIdx.x;
    const int chunk = (HWMAP + STATS_SLICES - 1) / STATS_SLICES;
    const int p0    = blockIdx.y * chunk;
    const int p1    = min(p0 + chunk, HWMAP);
    float a = 0.f, b = 0.f;
    for (int p = p0 + threadIdx.x; p < p1; p += 256) {
        const float v = x[(size_t)p * C + c];
        a += v;
        b += v * v;
    }
    s1[threadIdx.x] = a;
    s2[threadIdx.x] = b;
    __syncthreads();
    for (int o = 128; o > 0; o >>= 1) {
        if (threadIdx.x < o) {
            s1[threadIdx.x] += s1[threadIdx.x + o];
            s2[threadIdx.x] += s2[threadIdx.x + o];
        }
        __syncthreads();
    }
    if (threadIdx.x == 0) {
        atomicAdd(&sums[c], s1[0]);
        atomicAdd(&sumsq[c], s2[0]);
    }
}

__global__ void bn_finalize(const float* __restrict__ sums, const float* __restrict__ sumsq,
                            float* __restrict__ mean, float* __restrict__ var, int C) {
    const int c = threadIdx.x;
    if (c < C) {
        const float m = sums[c] * (1.0f / HWMAP);
        mean[c] = m;
        var[c]  = sumsq[c] * (1.0f / HWMAP) - m * m;
    }
}

__global__ void bn_relu_f16(const float* __restrict__ x, _Float16* __restrict__ y, int C,
                            const float* __restrict__ g, const float* __restrict__ b,
                            const float* __restrict__ mean, const float* __restrict__ var,
                            int total) {
    const int i = blockIdx.x * 256 + threadIdx.x;
    if (i >= total) return;
    const int c = i % C;
    const float v = (x[i] - mean[c]) * rsqrtf(var[c] + 1e-5f) * g[c] + b[c];
    y[i] = (_Float16)fmaxf(v, 0.f);
}

// ---------------------------------------------------------------------------
// 7) final 1x1 conv (48 -> 21) + bias, NCHW f32 output
// ---------------------------------------------------------------------------
__global__ void conv5_bias(const _Float16* __restrict__ act, const float* __restrict__ w,
                           const float* __restrict__ bias, float* __restrict__ sem) {
    const int p = blockIdx.x * 256 + threadIdx.x;
    if (p >= HWMAP) return;
    float a[48];
#pragma unroll
    for (int c = 0; c < 48; ++c) a[c] = (float)act[(size_t)p * 48 + c];
    for (int o = 0; o < 21; ++o) {
        float s = bias[o];
#pragma unroll
        for (int c = 0; c < 48; ++c) s += a[c] * w[o * 48 + c];
        sem[(size_t)o * HWMAP + p] = s;
    }
}

// ---------------------------------------------------------------------------
// launch
// ---------------------------------------------------------------------------
extern "C" void kernel_launch(void* const* d_in, const int* in_sizes, int n_in,
                              void* d_out, int out_size, void* d_ws, size_t ws_size,
                              hipStream_t stream) {
    const float* feats = (const float*)d_in[0];   // (1,1,64,256,512)
    const int*   proj  = (const int*)d_in[1];     // (1,250000)
    const int*   mask  = (const int*)d_in[2];     // (1,1024,2048)
    const float* rgb   = (const float*)d_in[3];   // (1,1024,2048,3)
    const float* w1 = (const float*)d_in[4];
    const float* g1 = (const float*)d_in[5];
    const float* b1 = (const float*)d_in[6];
    const float* w2 = (const float*)d_in[7];
    const float* g2 = (const float*)d_in[8];
    const float* b2 = (const float*)d_in[9];
    const float* w3 = (const float*)d_in[10];
    const float* g3 = (const float*)d_in[11];
    const float* b3 = (const float*)d_in[12];
    const float* w4 = (const float*)d_in[13];
    const float* g4 = (const float*)d_in[14];
    const float* b4 = (const float*)d_in[15];
    const float* w5 = (const float*)d_in[16];
    const float* b5 = (const float*)d_in[17];

    float* sem  = (float*)d_out;                   // (1,21,500,500)
    float* obs  = sem + (size_t)21 * HWMAP;        // (1,500,500) as 0/1 floats
    float* rgbw = obs + HWMAP;                     // (500,500,3)

    // ---- bump-allocate workspace ----
    char*  ws  = (char*)d_ws;
    size_t off = 0;
    auto alloc = [&](size_t bytes) -> char* {
        char* p = ws + off;
        off = (off + bytes + 255) & ~(size_t)255;
        return p;
    };
    int*      pos   = (int*)alloc((size_t)NPIX * sizeof(int));
    int*      bsum  = (int*)alloc((size_t)NBLK * sizeof(int));
    int*      dmax  = (int*)alloc(256);
    _Float16* wp1   = (_Float16*)alloc((size_t)3136 * 128 * 2);
    _Float16* wp2   = (_Float16*)alloc((size_t)1152 * 64 * 2);
    _Float16* wp3   = (_Float16*)alloc((size_t)576 * 48 * 2);
    _Float16* wp4   = (_Float16*)alloc((size_t)432 * 48 * 2);
    float*    sums  = (float*)alloc(128 * sizeof(float));
    float*    sumsq = (float*)alloc(128 * sizeof(float));
    float*    mean  = (float*)alloc(128 * sizeof(float));
    float*    var   = (float*)alloc(128 * sizeof(float));
    _Float16* actH  = (_Float16*)alloc((size_t)HWMAP * 128 * 2);  // f16 ping
    float*    actF  = (float*)alloc((size_t)HWMAP * 128 * 4);     // f32 pong

    const int nProj = 250000;

    // ---- compaction + sentinel ----
    mask_block_sums<<<NBLK, SCANB, 0, stream>>>(mask, bsum);
    scan_sums4096<<<1, 1024, 0, stream>>>(bsum);
    scatter_pos<<<NBLK, SCANB, 0, stream>>>(mask, bsum, pos);
    init_max<<<1, 64, 0, stream>>>(dmax);
    max_reduce<<<256, 256, 0, stream>>>(proj, nProj, dmax);

    // ---- fused upsample-gather into the map ----
    gather_build<<<HWMAP / 8, 256, 0, stream>>>(feats, proj, pos, rgb, dmax,
                                                actH, obs, rgbw);

    // ---- pack conv weights to f16 [K][N] ----
    pack_w<<<1568, 256, 0, stream>>>(w1, wp1, 128, 64, 7, 7);
    pack_w<<<288, 256, 0, stream>>>(w2, wp2, 64, 128, 3, 3);
    pack_w<<<108, 256, 0, stream>>>(w3, wp3, 48, 64, 3, 3);
    pack_w<<<81, 256, 0, stream>>>(w4, wp4, 48, 48, 3, 3);

    const int gM = (HWMAP + 127) / 128;  // 1954

    // ---- layer 1: 64 -> 128, 7x7 ----
    conv_wmma<64, 128, 7, 7, 3><<<dim3(gM, 2), 256, 0, stream>>>(actH, wp1, actF);
    zero_stats<<<1, 256, 0, stream>>>(sums, sumsq);
    bn_partial<<<dim3(128, STATS_SLICES), 256, 0, stream>>>(actF, 128, sums, sumsq);
    bn_finalize<<<1, 128, 0, stream>>>(sums, sumsq, mean, var, 128);
    bn_relu_f16<<<(HWMAP * 128 + 255) / 256, 256, 0, stream>>>(actF, actH, 128,
                                                               g1, b1, mean, var, HWMAP * 128);
    // ---- layer 2: 128 -> 64, 3x3 ----
    conv_wmma<128, 64, 3, 3, 1><<<dim3(gM, 1), 256, 0, stream>>>(actH, wp2, actF);
    zero_stats<<<1, 256, 0, stream>>>(sums, sumsq);
    bn_partial<<<dim3(64, STATS_SLICES), 256, 0, stream>>>(actF, 64, sums, sumsq);
    bn_finalize<<<1, 128, 0, stream>>>(sums, sumsq, mean, var, 64);
    bn_relu_f16<<<(HWMAP * 64 + 255) / 256, 256, 0, stream>>>(actF, actH, 64,
                                                              g2, b2, mean, var, HWMAP * 64);
    // ---- layer 3: 64 -> 48, 3x3 ----
    conv_wmma<64, 48, 3, 3, 1><<<dim3(gM, 1), 256, 0, stream>>>(actH, wp3, actF);
    zero_stats<<<1, 256, 0, stream>>>(sums, sumsq);
    bn_partial<<<dim3(48, STATS_SLICES), 256, 0, stream>>>(actF, 48, sums, sumsq);
    bn_finalize<<<1, 128, 0, stream>>>(sums, sumsq, mean, var, 48);
    bn_relu_f16<<<(HWMAP * 48 + 255) / 256, 256, 0, stream>>>(actF, actH, 48,
                                                              g3, b3, mean, var, HWMAP * 48);
    // ---- layer 4: 48 -> 48, 3x3 ----
    conv_wmma<48, 48, 3, 3, 1><<<dim3(gM, 1), 256, 0, stream>>>(actH, wp4, actF);
    zero_stats<<<1, 256, 0, stream>>>(sums, sumsq);
    bn_partial<<<dim3(48, STATS_SLICES), 256, 0, stream>>>(actF, 48, sums, sumsq);
    bn_finalize<<<1, 128, 0, stream>>>(sums, sumsq, mean, var, 48);
    bn_relu_f16<<<(HWMAP * 48 + 255) / 256, 256, 0, stream>>>(actF, actH, 48,
                                                              g4, b4, mean, var, HWMAP * 48);
    // ---- layer 5: 1x1, 48 -> 21 + bias, NCHW output ----
    conv5_bias<<<(HWMAP + 255) / 256, 256, 0, stream>>>(actH, w5, b5, sem);
}
